// QuatBlockSSMCuda_17102559773346
// MI455X (gfx1250) — compile-verified, matrix operand order checked by reading, not run on hardware
//
#include <hip/hip_runtime.h>
#include <hip/hip_bf16.h>
#include <math.h>

// ---------------- problem constants ----------------
#define BATCH 8
#define TLEN  4096
#define DDIM  768
#define NBQ   256           // quaternion blocks (DDIM/3)
#define MROWS (BATCH * TLEN)  // 32768
#define NCOLS (2 * DDIM)      // 1536 : [0,768)=biv, [768,1536)=inj
#define KDIM  DDIM

// ---------------- GEMM tiling ----------------
#define BM 128
#define BN 128
#define BK 32
#define BKP (BK + 8)        // padded LDS row stride (bf16 elems) for bank spread

typedef __attribute__((ext_vector_type(16))) __bf16 v16bf;
typedef __attribute__((ext_vector_type(8)))  __bf16 v8bf;
typedef __attribute__((ext_vector_type(8)))  float  v8f;

// Load one 16x32 bf16 A/B fragment from an LDS plane laid out [row][BKP].
// CDNA5 16-bit operand layout: lanes 0-15 hold row=lane, K={0..7,16..23};
// lanes 16-31 hold row=lane-16, K={8..15,24..31}. Two aligned b128 loads.
__device__ __forceinline__ v16bf load_frag(const __bf16* plane, int row_base, int lane) {
  const int row   = row_base + (lane & 15);
  const int khalf = (lane >> 4) * 8;
  const __bf16* p = plane + row * BKP + khalf;
  v8bf c0 = *(const v8bf*)(p);
  v8bf c1 = *(const v8bf*)(p + 16);
  v16bf r;
#pragma unroll
  for (int i = 0; i < 8; ++i) { r[i] = c0[i]; r[i + 8] = c1[i]; }
  return r;
}

// Fused projection GEMM: out[m, n] for n in [0,1536).
//   n <  768 : biv column n      (weights W_biv, bias b_biv)  -> ws biv buffer
//   n >= 768 : inj column n-768  (weights W_in,  bias b_in)   -> d_out (in-place scanned later)
__global__ __launch_bounds__(256)
void qssm_proj_gemm(const float* __restrict__ X,
                    const float* __restrict__ Wbiv,
                    const float* __restrict__ Win,
                    const float* __restrict__ bbiv,
                    const float* __restrict__ bin,
                    float* __restrict__ out_biv,
                    float* __restrict__ out_inj) {
  __shared__ __bf16 As_hi[BM * BKP];
  __shared__ __bf16 As_lo[BM * BKP];
  __shared__ __bf16 Bs_hi[BN * BKP];
  __shared__ __bf16 Bs_lo[BN * BKP];

  const int tid  = threadIdx.x;
  const int lane = tid & 31;
  const int wave = tid >> 5;      // 0..7
  const int wm   = wave & 3;      // 4 waves along M (32 rows each)
  const int wn   = wave >> 2;     // 2 waves along N (64 cols each)

  const int m0 = blockIdx.x * BM;
  const int n0 = blockIdx.y * BN;

  v8f acc[2][4];
#pragma unroll
  for (int i = 0; i < 2; ++i)
#pragma unroll
    for (int j = 0; j < 4; ++j) acc[i][j] = (v8f)0.0f;

  // staging map: 8 threads per row (8 x float4 = 32 floats), 32 rows per pass
  const int srow = tid >> 3;        // 0..31
  const int scol = (tid & 7) * 4;   // 0,4,...,28

  for (int kt = 0; kt < KDIM / BK; ++kt) {
    const int k0 = kt * BK;

    // ---- stage A (x tile) and B (weight tile) into LDS as bf16 hi/lo planes ----
#pragma unroll
    for (int pass = 0; pass < 4; ++pass) {
      const int row = pass * 32 + srow;

      // A: x[m0+row, k0+scol .. +3]
      const float* pa = X + (size_t)(m0 + row) * KDIM + k0 + scol;
      float4 va = *(const float4*)pa;
      if (k0 + BK < KDIM) __builtin_prefetch(pa + BK, 0, 1);
#pragma unroll
      for (int q = 0; q < 4; ++q) {
        float f   = ((const float*)&va)[q];
        __bf16 h  = (__bf16)f;
        __bf16 l  = (__bf16)(f - (float)h);
        As_hi[row * BKP + scol + q] = h;
        As_lo[row * BKP + scol + q] = l;
      }

      // B: weight row (n0+row), cols k0+scol..+3 ; pick W_biv vs W_in
      const int ng = n0 + row;
      const float* wrow = (ng < DDIM) ? (Wbiv + (size_t)ng * KDIM)
                                      : (Win + (size_t)(ng - DDIM) * KDIM);
      const float* pb = wrow + k0 + scol;
      float4 vb = *(const float4*)pb;
      if (k0 + BK < KDIM) __builtin_prefetch(pb + BK, 0, 1);
#pragma unroll
      for (int q = 0; q < 4; ++q) {
        float f   = ((const float*)&vb)[q];
        __bf16 h  = (__bf16)f;
        __bf16 l  = (__bf16)(f - (float)h);
        Bs_hi[row * BKP + scol + q] = h;
        Bs_lo[row * BKP + scol + q] = l;
      }
    }
    __syncthreads();

    // ---- fragments ----
    v16bf a_hi[2], a_lo[2], b_hi[4], b_lo[4];
#pragma unroll
    for (int i = 0; i < 2; ++i) {
      a_hi[i] = load_frag(As_hi, wm * 32 + i * 16, lane);
      a_lo[i] = load_frag(As_lo, wm * 32 + i * 16, lane);
    }
#pragma unroll
    for (int j = 0; j < 4; ++j) {
      b_hi[j] = load_frag(Bs_hi, wn * 64 + j * 16, lane);
      b_lo[j] = load_frag(Bs_lo, wn * 64 + j * 16, lane);
    }

    // ---- bf16x3 split-precision WMMA: acc += Ah*Bh + Ah*Bl + Al*Bh ----
#pragma unroll
    for (int i = 0; i < 2; ++i) {
#pragma unroll
      for (int j = 0; j < 4; ++j) {
        acc[i][j] = __builtin_amdgcn_wmma_f32_16x16x32_bf16(
            false, a_hi[i], false, b_hi[j], (short)0, acc[i][j], false, false);
        acc[i][j] = __builtin_amdgcn_wmma_f32_16x16x32_bf16(
            false, a_hi[i], false, b_lo[j], (short)0, acc[i][j], false, false);
        acc[i][j] = __builtin_amdgcn_wmma_f32_16x16x32_bf16(
            false, a_lo[i], false, b_hi[j], (short)0, acc[i][j], false, false);
      }
    }
    __syncthreads();
  }

  // ---- epilogue: bias + scatter to biv workspace or inj (=d_out) ----
  // C/D layout: VGPR r, lanes 0-15 -> M=r, N=lane; lanes 16-31 -> M=8+r, N=lane-16.
  const int half = (lane >> 4) & 1;
#pragma unroll
  for (int i = 0; i < 2; ++i) {
#pragma unroll
    for (int j = 0; j < 4; ++j) {
      const int mb = m0 + wm * 32 + i * 16 + half * 8;
      const int nn = n0 + wn * 64 + j * 16 + (lane & 15);
      float bias;
      float* dst;
      int col;
      if (nn < DDIM) { bias = bbiv[nn]; dst = out_biv; col = nn; }
      else           { bias = bin[nn - DDIM]; dst = out_inj; col = nn - DDIM; }
#pragma unroll
      for (int r = 0; r < 8; ++r) {
        dst[(size_t)(mb + r) * DDIM + col] = acc[i][j][r] + bias;
      }
    }
  }
}

// decay[m] = sigmoid(dot(x[m,:], W_decay) + b_decay); one wave per row.
__global__ __launch_bounds__(256)
void qssm_decay(const float* __restrict__ X,
                const float* __restrict__ Wd,
                const float* __restrict__ bd,
                float* __restrict__ dec) {
  const int row  = blockIdx.x * 8 + (threadIdx.x >> 5);
  const int lane = threadIdx.x & 31;
  const float* xr = X + (size_t)row * DDIM;
  float s = 0.0f;
#pragma unroll 4
  for (int k = lane; k < DDIM; k += 32) s += xr[k] * Wd[k];
#pragma unroll
  for (int off = 16; off > 0; off >>= 1) s += __shfl_xor(s, off, 32);
  if (lane == 0) {
    float z = s + bd[0];
    dec[row] = 1.0f / (1.0f + expf(-z));
  }
}

// Sequential Rodrigues scan: one thread per (b, nb) chain; inj lives in `out`
// and is overwritten in place with the state h_t.
__global__ __launch_bounds__(256)
void qssm_scan(const float* __restrict__ biv,
               const float* __restrict__ dec,
               float* __restrict__ out) {
  const int b  = blockIdx.x;
  const int nb = threadIdx.x;  // 0..255
  float hx = 0.0f, hy = 0.0f, hz = 0.0f;
  const size_t base = ((size_t)b * TLEN) * DDIM + (size_t)nb * 3;
  const float* decb = dec + (size_t)b * TLEN;

  for (int t = 0; t < TLEN; ++t) {
    const size_t idx = base + (size_t)t * DDIM;
    const float bx = biv[idx], by = biv[idx + 1], bz = biv[idx + 2];
    const float ux = out[idx], uy = out[idx + 1], uz = out[idx + 2];
    const float d  = decb[t];

    const float th  = sqrtf(bx * bx + by * by + bz * bz);
    const float inv = 1.0f / fmaxf(th, 1e-8f);
    const float kx = bx * inv, ky = by * inv, kz = bz * inv;
    const float c = cosf(th), s = sinf(th);

    const float cx = ky * hz - kz * hy;
    const float cy = kz * hx - kx * hz;
    const float cz = kx * hy - ky * hx;
    const float kdoth = kx * hx + ky * hy + kz * hz;
    const float om = (1.0f - c) * kdoth;

    const float rx = hx * c + cx * s + kx * om;
    const float ry = hy * c + cy * s + ky * om;
    const float rz = hz * c + cz * s + kz * om;

    hx = d * rx + ux;
    hy = d * ry + uy;
    hz = d * rz + uz;

    out[idx]     = hx;
    out[idx + 1] = hy;
    out[idx + 2] = hz;
  }
}

extern "C" void kernel_launch(void* const* d_in, const int* in_sizes, int n_in,
                              void* d_out, int out_size, void* d_ws, size_t ws_size,
                              hipStream_t stream) {
  const float* x      = (const float*)d_in[0];  // (B,T,D)
  const float* W_biv  = (const float*)d_in[1];  // (768,768)
  const float* b_biv  = (const float*)d_in[2];  // (768,)
  const float* W_dec  = (const float*)d_in[3];  // (1,768)
  const float* b_dec  = (const float*)d_in[4];  // (1,)
  const float* W_in   = (const float*)d_in[5];  // (768,768)
  const float* b_in   = (const float*)d_in[6];  // (768,)
  float* out = (float*)d_out;                   // (B,T,768)

  float* ws_biv = (float*)d_ws;                       // MROWS*768 floats
  float* ws_dec = ws_biv + (size_t)MROWS * DDIM;      // MROWS floats

  // 1) fused projections: biv -> ws, inj -> d_out (in place for the scan)
  dim3 ggrid(MROWS / BM, NCOLS / BN);  // (256, 12)
  qssm_proj_gemm<<<ggrid, 256, 0, stream>>>(x, W_biv, W_in, b_biv, b_in,
                                            ws_biv, out);
  // 2) decay gate (GEMV + sigmoid)
  qssm_decay<<<MROWS / 8, 256, 0, stream>>>(x, W_dec, b_dec, ws_dec);
  // 3) sequential Rodrigues scan over T, one thread per (b, nb) chain
  qssm_scan<<<BATCH, NBQ, 0, stream>>>(ws_biv, ws_dec, out);
}
